// MultiViewEncoder_62088047231305
// MI455X (gfx1250) — matrix operand order repaired and checked by reading, not compiled, not run on hardware
//
#include <hip/hip_runtime.h>
#include <hip/hip_bf16.h>

typedef __attribute__((ext_vector_type(16))) _Float16 v16h;
typedef __attribute__((ext_vector_type(8)))  float    v8f;

#define BS     4
#define NV     12
#define CIN    384
#define COUT   32
#define HW     4096            // 64*64 feature map
#define NVOX   (64*64*64)
#define KSTEPS 12              // CIN / 32
#define NFRAG  (KSTEPS*2*32*16)  // 12288 halves of pre-swizzled W

// ws layout: [0, 24576): f16 A-fragments; [32768, +BS*NV*HW*COUT*4): R buffer [b][v][hw][c]
#define WS_R_OFF 32768

// ---------------------------------------------------------------------------
// K0: pre-swizzle w_reduce (COUT x CIN, f32) into WMMA f16 A-fragment order.
// A-matrix 16x32 f16 layout (ISA 7.12.2): lane L holds row m=L&15;
// half h of v16h holds K = kstep*32 + ((h<8) ? base+h : 16+base+(h-8)),
// base = (L>>4)*8.  Fragment index = (kstep*2 + mtile)*32 + lane.
// ---------------------------------------------------------------------------
__global__ void mv_prep_wfrag(const float* __restrict__ W,
                              _Float16* __restrict__ frag) {
    int idx = blockIdx.x * 256 + threadIdx.x;
    if (idx >= NFRAG) return;
    int h     = idx & 15;
    int lane  = (idx >> 4) & 31;
    int mtile = (idx >> 9) & 1;
    int kstep = idx >> 10;
    int m     = mtile * 16 + (lane & 15);
    int base  = (lane >> 4) * 8;
    int kl    = (h < 8) ? (base + h) : (16 + base + (h - 8));
    frag[idx] = (_Float16)W[m * CIN + kstep * 32 + kl];
}

// ---------------------------------------------------------------------------
// K1: per-view GEMM  R[bv][hw][c] = W(32x384) @ F(384x4096) + bias, via
// v_wmma_f32_16x16x32_f16.  Block = 4 waves; each wave does 4 column tiles
// of 16 pixels, both M-tiles (cout 0..15, 16..31).  A fragments come from
// LDS; B fragments are built from global f32 loads converted to f16.
// B-matrix 32x16 f16 layout: lane gives n=lane&15; lanes 0-15 hold K rows
// kstep*32+0..7,16..23, lanes 16-31 hold kstep*32+8..15,24..31 (same K order as A).
// ---------------------------------------------------------------------------
__global__ __launch_bounds__(128) void mv_reduce_gemm(
    const float* __restrict__ F,          // vit_feats, (BS*NV, CIN, HW)
    const float* __restrict__ bias,       // (COUT)
    const _Float16* __restrict__ wfrag,   // pre-swizzled A fragments
    float* __restrict__ R)                // (BS*NV, HW, COUT)
{
    __shared__ __align__(32) _Float16 lA[NFRAG];

    // cooperative LDS fill of the A fragments (24576 B = 1536 uint4)
    {
        const uint4* src = (const uint4*)wfrag;
        uint4*       dst = (uint4*)lA;
        for (int i = threadIdx.x; i < (int)(NFRAG * sizeof(_Float16) / 16); i += 128)
            dst[i] = src[i];
    }
    __syncthreads();

    const int wave   = threadIdx.x >> 5;
    const int lane   = threadIdx.x & 31;
    const int bv     = blockIdx.y;
    const int nlo    = lane & 15;
    const int laneHi = lane >> 4;
    const int khalf  = laneHi << 4;       // 0 or 16: which half of the K=32 slab
    const float* Fv  = F + (size_t)bv * CIN * HW;

    for (int t = 0; t < 4; ++t) {
        const int ncol0 = (blockIdx.x * 16 + wave * 4 + t) * 16;
        const int n     = ncol0 + nlo;
        v8f acc0 = {};
        v8f acc1 = {};

        for (int ks = 0; ks < KSTEPS; ++ks) {
            const int kb = ks * 32 + khalf;
            // build B fragment: 16 K-rows of column n, f32 -> f16
            v16h bf;
            #pragma unroll
            for (int h = 0; h < 16; ++h)
                bf[h] = (_Float16)Fv[(size_t)(ks * 32 + ((h < 8) ? (laneHi * 8 + h) : (16 + laneHi * 8 + (h - 8)))) * HW + n];
            // prefetch next K slab (global_prefetch_b8)
            if (ks + 1 < KSTEPS)
                __builtin_prefetch(&Fv[(size_t)(kb + 32) * HW + n], 0, 1);

            const v16h a0 = *(const v16h*)&lA[((ks * 2 + 0) * 32 + lane) * 16];
            const v16h a1 = *(const v16h*)&lA[((ks * 2 + 1) * 32 + lane) * 16];

            acc0 = __builtin_amdgcn_wmma_f32_16x16x32_f16(
                       false, a0, false, bf, (short)0, acc0, false, false);
            acc1 = __builtin_amdgcn_wmma_f32_16x16x32_f16(
                       false, a1, false, bf, (short)0, acc1, false, false);
        }

        // Epilogue: D layout (ISA 7.12.2): lane col = n, VGPR e holds
        // m = e + laneHi*8 (+16 for mtile 1).  Fuse bias, store channel-
        // contiguous so K2 gathers are 128B-contiguous per pixel.
        float* outp = R + ((size_t)bv * HW + n) * COUT;
        const float4* bp0 = (const float4*)(bias + laneHi * 8);
        const float4* bp1 = (const float4*)(bias + 16 + laneHi * 8);
        float4 b00 = bp0[0], b01 = bp0[1], b10 = bp1[0], b11 = bp1[1];

        float4 v00 = make_float4(acc0[0] + b00.x, acc0[1] + b00.y,
                                 acc0[2] + b00.z, acc0[3] + b00.w);
        float4 v01 = make_float4(acc0[4] + b01.x, acc0[5] + b01.y,
                                 acc0[6] + b01.z, acc0[7] + b01.w);
        float4 v10 = make_float4(acc1[0] + b10.x, acc1[1] + b10.y,
                                 acc1[2] + b10.z, acc1[3] + b10.w);
        float4 v11 = make_float4(acc1[4] + b11.x, acc1[5] + b11.y,
                                 acc1[6] + b11.z, acc1[7] + b11.w);

        float* p0 = outp + laneHi * 8;        // mtile 0
        *(float4*)(p0)     = v00;
        *(float4*)(p0 + 4) = v01;
        float* p1 = outp + 16 + laneHi * 8;   // mtile 1
        *(float4*)(p1)     = v10;
        *(float4*)(p1 + 4) = v11;
    }
}

// ---------------------------------------------------------------------------
// K2: backproject + average.  One thread per (b, voxel).  Loop views,
// project (proj rows 0,1 pre-divided by STRIDE=4), round-half-even like
// jnp.round, gather 32 contiguous channel floats (8 x b128) when valid,
// normalize by valid count, coalesced store in (b, cout, N) layout.
// ---------------------------------------------------------------------------
__global__ __launch_bounds__(256) void mv_backproject(
    const float* __restrict__ R,       // (BS*NV, HW, COUT)
    const float* __restrict__ proj,    // (BS, NV, 3, 4)
    float* __restrict__ out)           // (BS, COUT, NVOX)
{
    const int vox = blockIdx.x * 256 + threadIdx.x;
    const int b   = blockIdx.y;
    const int x   = vox >> 12;
    const int y   = (vox >> 6) & 63;
    const int z   = vox & 63;
    const float wx = x * 0.04f - 1.28f;
    const float wy = y * 0.04f - 1.28f;
    const float wz = z * 0.04f - 1.28f;

    float acc[COUT];
    #pragma unroll
    for (int c = 0; c < COUT; ++c) acc[c] = 0.0f;
    float cnt = 0.0f;

    for (int v = 0; v < NV; ++v) {
        const float* P = proj + (size_t)(b * NV + v) * 12;
        float cz = P[8] * wx + P[9] * wy + P[10] * wz + P[11];
        float cx = (P[0] * wx + P[1] * wy + P[2] * wz + P[3]) * 0.25f;
        float cy = (P[4] * wx + P[5] * wy + P[6] * wz + P[7]) * 0.25f;
        float rz = 1.0f / cz;
        int px = (int)__builtin_rintf(cx * rz);
        int py = (int)__builtin_rintf(cy * rz);
        bool valid = (px >= 0) & (py >= 0) & (px < 64) & (py < 64) & (cz > 0.0f);
        if (valid) {
            cnt += 1.0f;
            const float4* rv = (const float4*)(
                R + ((size_t)(b * NV + v) * HW + (py * 64 + px)) * COUT);
            #pragma unroll
            for (int q = 0; q < 8; ++q) {
                float4 f = rv[q];
                acc[q * 4 + 0] += f.x;
                acc[q * 4 + 1] += f.y;
                acc[q * 4 + 2] += f.z;
                acc[q * 4 + 3] += f.w;
            }
        }
    }

    const float scale = (cnt > 0.0f) ? (1.0f / cnt) : 0.0f;
    float* ob = out + (size_t)b * COUT * NVOX + vox;
    #pragma unroll
    for (int c = 0; c < COUT; ++c)
        ob[(size_t)c * NVOX] = acc[c] * scale;
}

// ---------------------------------------------------------------------------
extern "C" void kernel_launch(void* const* d_in, const int* in_sizes, int n_in,
                              void* d_out, int out_size, void* d_ws, size_t ws_size,
                              hipStream_t stream) {
    const float* vit  = (const float*)d_in[0];   // (4,12,384,64,64) f32
    const float* wrd  = (const float*)d_in[1];   // (32,384) f32
    const float* brd  = (const float*)d_in[2];   // (32,) f32
    const float* proj = (const float*)d_in[3];   // (4,12,3,4) f32
    float* out        = (float*)d_out;           // (4,32,64,64,64) f32

    _Float16* wfrag = (_Float16*)d_ws;
    float*    R     = (float*)((char*)d_ws + WS_R_OFF);

    // K0: swizzle weights into WMMA A-fragment order
    mv_prep_wfrag<<<(NFRAG + 255) / 256, 256, 0, stream>>>(wrd, wfrag);

    // K1: 48 views x 4096 pixels; block covers 256 pixel columns
    dim3 g1(HW / 256, BS * NV);
    mv_reduce_gemm<<<g1, 128, 0, stream>>>(vit, brd, wfrag, R);

    // K2: one thread per (b, voxel)
    dim3 g2(NVOX / 256, BS);
    mv_backproject<<<g2, 256, 0, stream>>>(R, proj, out);
}
